// MutualInformationLoss_68504728371273
// MI455X (gfx1250) — matrix operand order repaired and checked by reading, not compiled, not run on hardware
//
#include <hip/hip_runtime.h>
#include <hip/hip_bf16.h>
#include <math.h>

// Shapes from the reference
#define D_DIM   1024
#define L_DIM   256
#define NB      128     // B (number of sentences / graphs)
#define KCHUNK  32      // K per bf16 WMMA
#define NCHUNK  (D_DIM / KCHUNK)
#define BSTR    40      // padded LDS row stride in bf16 elems: 32 data + 8 pad (80 B, 16B-aligned)

typedef __attribute__((ext_vector_type(16))) __bf16 v16bf;
typedef __attribute__((ext_vector_type(8)))  float  v8f;

union frag16 { uint4 q[2]; unsigned w[8]; v16bf v; };

// Split a pair of f32 into packed bf16-hi (truncation) and bf16-lo (residual) dwords.
// hi = x & 0xFFFF0000 (exact truncation, reused as float for the residual),
// lo = bf16(x - hi).  hi+lo reproduces x to ~2^-16 relative.
__device__ __forceinline__ void split_pair(float x0, float x1, unsigned &hi_dw, unsigned &lo_dw) {
  const unsigned u0 = __float_as_uint(x0);
  const unsigned u1 = __float_as_uint(x1);
  const float h0 = __uint_as_float(u0 & 0xFFFF0000u);
  const float h1 = __uint_as_float(u1 & 0xFFFF0000u);
  hi_dw = (u1 & 0xFFFF0000u) | (u0 >> 16);       // -> v_perm_b32
  const __bf16 l0 = (__bf16)(x0 - h0);
  const __bf16 l1 = (__bf16)(x1 - h1);           // -> v_pk_add_f32 + v_cvt_pk_bf16_f32
  lo_dw = (unsigned)__builtin_bit_cast(unsigned short, l0) |
          ((unsigned)__builtin_bit_cast(unsigned short, l1) << 16);
}

// Fast softplus: hardware v_exp_f32 / v_log_f32 (vs ~100-instr ocml log1pf).
// |error| only matters for terms < 2^-24, invisible in the 8.4M-term sum.
__device__ __forceinline__ float softplus_f(float x) {
  return fmaxf(x, 0.0f) + __logf(1.0f + __expf(-fabsf(x)));
}

// One workgroup = 256 threads = 8 waves. Tile: 128 rows (M) x 128 cols (N=B).
// grid.x = 256 tiles (32768 rows / 128), grid.y = 2 (side 0 / side 1).
__global__ __launch_bounds__(256, 1)
void mi_gemm_kernel(const float* __restrict__ tok0, const float* __restrict__ tok1,
                    const float* __restrict__ glob0, const float* __restrict__ glob1,
                    const int* __restrict__ len0, const int* __restrict__ len1,
                    double* __restrict__ acc_out)
{
  // double-buffered glob K-chunk, pre-split to bf16: [buf][hi=0/lo=1][row][k]
  __shared__ unsigned short ldsb[2][2][NB * BSTR];   // 40 KB

  const int side = blockIdx.y;
  const float* tok  = side ? tok1  : tok0;
  const float* glob = side ? glob1 : glob0;
  const int*   len  = side ? len1  : len0;

  const int tile = blockIdx.x;            // each tile = 128 rows inside one sentence
  const int wave = threadIdx.x >> 5;      // 0..7
  const int lane = threadIdx.x & 31;

  const int b    = tile >> 1;             // sentence id owning this tile
  const int lenb = max(len[b], 1);        // clipped length

  const int row0 = tile * 128 + wave * 16;  // this wave's 16-row strip

  // A-fragment addressing (16x32 bf16 ISA layout):
  // lanes 0-15: M=lane, K = {0..7, 16..23}; lanes 16-31: M=lane-16, K = {8..15, 24..31}
  const int am = lane & 15;
  const int ak = (lane >> 4) * 8;
  const float* arow = tok + (size_t)(row0 + am) * D_DIM;

  // B-fragment addressing: lanes 0-15: N=lane, K = {0..7, 16..23}; lanes 16-31: N=lane-16, K = {8..15, 24..31}
  const int bg = lane & 15;
  const int bk = (lane >> 4) * 8;

  v8f acc[8] = {};   // 8 N-subtiles of 16x16 f32 accumulators

  // cooperative fill mapping: 256 threads convert 128 rows x 32 floats -> bf16 hi/lo
  const int fg = threadIdx.x >> 1;         // glob row 0..127
  const int fh = (threadIdx.x & 1) * 16;   // k offset 0 or 16

  auto fill = [&](int buf, int kc) {
    const float4* src = (const float4*)(glob + (size_t)fg * D_DIM + kc + fh);
    float f[16];
    {
      const float4 s0 = src[0], s1 = src[1], s2 = src[2], s3 = src[3];
      f[0]=s0.x;  f[1]=s0.y;  f[2]=s0.z;  f[3]=s0.w;
      f[4]=s1.x;  f[5]=s1.y;  f[6]=s1.z;  f[7]=s1.w;
      f[8]=s2.x;  f[9]=s2.y;  f[10]=s2.z; f[11]=s2.w;
      f[12]=s3.x; f[13]=s3.y; f[14]=s3.z; f[15]=s3.w;
    }
    unsigned hi[8], lo[8];
    #pragma unroll
    for (int p = 0; p < 8; ++p) split_pair(f[2*p], f[2*p+1], hi[p], lo[p]);
    uint4* dh = (uint4*)&ldsb[buf][0][fg * BSTR + fh];
    uint4* dl = (uint4*)&ldsb[buf][1][fg * BSTR + fh];
    dh[0] = make_uint4(hi[0], hi[1], hi[2], hi[3]);
    dh[1] = make_uint4(hi[4], hi[5], hi[6], hi[7]);
    dl[0] = make_uint4(lo[0], lo[1], lo[2], lo[3]);
    dl[1] = make_uint4(lo[4], lo[5], lo[6], lo[7]);
  };

  fill(0, 0);
  __syncthreads();

  for (int c = 0; c < NCHUNK; ++c) {
    const int kc  = c * KCHUNK;
    const int buf = c & 1;

    // ---- A loads for the current chunk (issue early) ----
    const float4 a0 = *(const float4*)(arow + kc + ak);
    const float4 a1 = *(const float4*)(arow + kc + ak + 4);
    const float4 a2 = *(const float4*)(arow + kc + 16 + ak);
    const float4 a3 = *(const float4*)(arow + kc + 16 + ak + 4);

    // ---- stage+convert the next glob chunk into the other buffer ----
    if (c + 1 < NCHUNK) fill(buf ^ 1, kc + KCHUNK);

    // keep the private A stream ahead of the consumers (global_prefetch_b8)
    if (kc + 2 * KCHUNK < D_DIM) __builtin_prefetch(arow + kc + 2 * KCHUNK, 0, 1);

    // ---- split A into bf16 hi/lo fragments ----
    float af[16];
    af[0]=a0.x;  af[1]=a0.y;  af[2]=a0.z;  af[3]=a0.w;
    af[4]=a1.x;  af[5]=a1.y;  af[6]=a1.z;  af[7]=a1.w;
    af[8]=a2.x;  af[9]=a2.y;  af[10]=a2.z; af[11]=a2.w;
    af[12]=a3.x; af[13]=a3.y; af[14]=a3.z; af[15]=a3.w;
    frag16 ah, al;
    #pragma unroll
    for (int p = 0; p < 8; ++p) split_pair(af[2*p], af[2*p+1], ah.w[p], al.w[p]);

    // ---- 8 N-subtiles: pre-split B straight from LDS, 3 WMMAs each ----
    #pragma unroll
    for (int n = 0; n < 8; ++n) {
      const int roff = (n * 16 + bg) * BSTR + bk;
      const uint4* bh = (const uint4*)&ldsb[buf][0][roff];
      const uint4* bl = (const uint4*)&ldsb[buf][1][roff];
      frag16 vh, vl;
      vh.q[0] = bh[0]; vh.q[1] = bh[2];
      vl.q[0] = bl[0]; vl.q[1] = bl[2];

      acc[n] = __builtin_amdgcn_wmma_f32_16x16x32_bf16(false, ah.v, false, vh.v, (short)0, acc[n], false, false);
      acc[n] = __builtin_amdgcn_wmma_f32_16x16x32_bf16(false, ah.v, false, vl.v, (short)0, acc[n], false, false);
      acc[n] = __builtin_amdgcn_wmma_f32_16x16x32_bf16(false, al.v, false, vh.v, (short)0, acc[n], false, false);
    }
    __syncthreads();
  }

  // ---- fused epilogue: masked softplus reduction ----
  // C layout: VGPR j -> M=j (lanes 0-15) / M=j+8 (lanes 16-31), N = n*16 + (lane&15)
  float sp_pos = 0.0f, sp_neg = 0.0f;
  const int chalf = lane >> 4;
  const int lbase = (tile & 1) * 128 + wave * 16 + 8 * chalf;
  #pragma unroll
  for (int n = 0; n < 8; ++n) {
    const int g = n * 16 + (lane & 15);
    #pragma unroll
    for (int j = 0; j < 8; ++j) {
      const int l = lbase + j;               // token index within sentence
      if (l < lenb) {
        const float x = acc[n][j];
        if (g == b) sp_pos += softplus_f(-x);   // -> e_pos = -sum/nn
        else        sp_neg += softplus_f(x);    // softplus(-x)+x == softplus(x)
      }
    }
  }
  #pragma unroll
  for (int off = 16; off > 0; off >>= 1) {
    sp_pos += __shfl_down(sp_pos, off, 32);
    sp_neg += __shfl_down(sp_neg, off, 32);
  }
  if (lane == 0) {
    atomicAdd(&acc_out[0], (double)sp_pos);
    atomicAdd(&acc_out[1], (double)sp_neg);
  }
}

// Tiny finalize: length sums (clipped), loss, avg_sentence_length. One wave.
__global__ void mi_finalize_kernel(const int* __restrict__ len0, const int* __restrict__ len1,
                                   const double* __restrict__ acc, float* __restrict__ out)
{
  const int lane = threadIdx.x;   // 32 threads
  int s0 = 0, s1 = 0;
  for (int i = lane; i < NB; i += 32) { s0 += max(len0[i], 1); s1 += max(len1[i], 1); }
  #pragma unroll
  for (int off = 16; off > 0; off >>= 1) {
    s0 += __shfl_down(s0, off, 32);
    s1 += __shfl_down(s1, off, 32);
  }
  if (lane == 0) {
    const double nn    = (double)(s0 + s1);
    const double e_pos = -acc[0] / nn;
    const double e_neg =  acc[1] / (nn * (double)(NB - 1));
    out[0] = (float)(e_neg - e_pos);
    out[1] = (float)((((double)s0 / (double)NB) + ((double)s1 / (double)NB)) * 0.5);
  }
}

extern "C" void kernel_launch(void* const* d_in, const int* in_sizes, int n_in,
                              void* d_out, int out_size, void* d_ws, size_t ws_size,
                              hipStream_t stream) {
  const float* tok0  = (const float*)d_in[0];
  const float* tok1  = (const float*)d_in[1];
  const float* glob0 = (const float*)d_in[2];
  const float* glob1 = (const float*)d_in[3];
  const int*   len0  = (const int*)d_in[4];
  const int*   len1  = (const int*)d_in[5];
  float*  out = (float*)d_out;
  double* acc = (double*)d_ws;

  // zero the two f64 accumulators (capturable async memset)
  hipMemsetAsync(d_ws, 0, 2 * sizeof(double), stream);

  dim3 grid(256, 2);   // 256 M-tiles x 2 sides
  mi_gemm_kernel<<<grid, 256, 0, stream>>>(tok0, tok1, glob0, glob1, len0, len1, acc);
  mi_finalize_kernel<<<1, 32, 0, stream>>>(len0, len1, acc, out);
}
